// Layer_11776800325814
// MI455X (gfx1250) — compile-verified, matrix operand order checked by reading, not run on hardware
//
#include <hip/hip_runtime.h>
#include <hip/hip_bf16.h>

#define DMODEL 1024
#define DFF    4096
#define NEXP   8
#define NTOK   2048

// phase-1/2 Dff chunk per iteration and LDS row stride (bank-skewed)
#define FCH    256
#define SHSTR  264

typedef __attribute__((ext_vector_type(16))) __bf16 v16bf;
typedef __attribute__((ext_vector_type(8)))  float  v8f;

union BFrag {
    v16bf v;
    __bf16 b[16];
    uint4 q[2];
};

// Native conversion: lowers to v_cvt_pk_bf16_f32 (pairs merged by the backend)
__device__ __forceinline__ __bf16 f2bf(float f) { return (__bf16)f; }

// Fast SiLU: g * v_rcp_f32(1 + e^-g)
__device__ __forceinline__ float silu(float g) {
    return g * __builtin_amdgcn_rcpf(1.f + __expf(-g));
}

// ---------------------------------------------------------------------------
// Kernel 1: gating (top-2 softmax, renormalized) + fp32->bf16 copy of x.
// One wave (32 lanes) per token. 256 threads/block => 8 tokens/block.
// ---------------------------------------------------------------------------
__global__ __launch_bounds__(256) void moe_gate_kernel(
    const float* __restrict__ x, const float* __restrict__ Wg,
    int* __restrict__ counts, int* __restrict__ list,
    float* __restrict__ wlist, __bf16* __restrict__ xbf)
{
    const int wave = threadIdx.x >> 5;
    const int lane = threadIdx.x & 31;
    const int token = blockIdx.x * 8 + wave;
    if (token >= NTOK) return;

    const float* xr = x + (size_t)token * DMODEL;
    float acc[NEXP];
#pragma unroll
    for (int e = 0; e < NEXP; ++e) acc[e] = 0.f;

    for (int c = lane; c < DMODEL; c += 32) {
        float xv = xr[c];
        xbf[(size_t)token * DMODEL + c] = f2bf(xv);
        const float4* wr = (const float4*)(Wg + (size_t)c * NEXP);
        float4 a = wr[0], b = wr[1];
        acc[0] += xv * a.x; acc[1] += xv * a.y;
        acc[2] += xv * a.z; acc[3] += xv * a.w;
        acc[4] += xv * b.x; acc[5] += xv * b.y;
        acc[6] += xv * b.z; acc[7] += xv * b.w;
    }
#pragma unroll
    for (int off = 16; off > 0; off >>= 1) {
#pragma unroll
        for (int e = 0; e < NEXP; ++e)
            acc[e] += __shfl_xor(acc[e], off, 32);
    }

    if (lane == 0) {
        int i0 = 0; float v0 = acc[0];
#pragma unroll
        for (int e = 1; e < NEXP; ++e)
            if (acc[e] > v0) { v0 = acc[e]; i0 = e; }
        int i1 = (i0 == 0) ? 1 : 0; float v1 = acc[i1];
#pragma unroll
        for (int e = 0; e < NEXP; ++e)
            if (e != i0 && acc[e] > v1) { v1 = acc[e]; i1 = e; }
        // renormalized top-2 softmax: w0 = e^{l0}/(e^{l0}+e^{l1})
        float w0 = 1.f / (1.f + __expf(v1 - v0));
        float w1 = 1.f - w0;
        int p0 = atomicAdd(&counts[i0], 1);
        list[i0 * NTOK + p0] = token; wlist[i0 * NTOK + p0] = w0;
        int p1 = atomicAdd(&counts[i1], 1);
        list[i1 * NTOK + p1] = token; wlist[i1 * NTOK + p1] = w1;
    }
}

// ---------------------------------------------------------------------------
// Kernel 2: expert SwiGLU FFN over the routed token list, bf16 WMMA.
// Grid: (mtile, dff_partition, expert). Block: 512 threads = 16 waves.
// Block handles M=64 tokens, Dff slice of 1024 in chunks of F=256:
//   phase 1: wave w computes g/u for ntile=w (16 ff cols) x all 4 M-tiles
//            (each strided B-fragment feeds 4 WMMAs), bias+SiLU -> h in LDS
//   phase 2: y += h @ Wd_e slice (WMMA, K over chunk), accum in VGPRs
// Epilogue: out[token] += gate_w * (y + b_down)  via f32 global atomics.
// ---------------------------------------------------------------------------
__global__ __launch_bounds__(512, 1) void moe_ffn_kernel(
    const __bf16* __restrict__ xbf,
    const int* __restrict__ counts, const int* __restrict__ list,
    const float* __restrict__ wlist,
    const float* __restrict__ w_gate, const float* __restrict__ b_gate,
    const float* __restrict__ w_up,   const float* __restrict__ b_up,
    const float* __restrict__ w_down, const float* __restrict__ b_down,
    float* __restrict__ out)
{
    const int e    = blockIdx.z;
    const int mt   = blockIdx.x;          // 0..31 token tile of 64
    const int part = blockIdx.y;          // 0..3  Dff partition of 1024
    const int cnt  = counts[e];
    if (mt * 64 >= cnt) return;

    __shared__ __align__(16) __bf16 sh[64 * SHSTR]; // h chunk, padded rows
    __shared__ int   stok[64];
    __shared__ float swt[64];

    if (threadIdx.x < 64) {
        int idx = mt * 64 + (int)threadIdx.x;
        bool valid = idx < cnt;
        stok[threadIdx.x] = list[e * NTOK + (valid ? idx : mt * 64)];
        swt[threadIdx.x]  = valid ? wlist[e * NTOK + idx] : 0.f;
    }
    __syncthreads();

    const int wave = threadIdx.x >> 5;    // phase-1 ntile, phase-2 d-col group
    const int lane = threadIdx.x & 31;
    const int col  = lane & 15;           // M (A) / N (B,C) within a 16-tile
    const int hi   = lane >> 4;           // lane-half selects K/M halves

    const float* wg = w_gate + (size_t)e * DMODEL * DFF;
    const float* wu = w_up   + (size_t)e * DMODEL * DFF;
    const float* wd = w_down + (size_t)e * DFF * DMODEL;

    const __bf16* arow[4];
#pragma unroll
    for (int m = 0; m < 4; ++m)
        arow[m] = xbf + (size_t)stok[m * 16 + col] * DMODEL;

    v8f yacc[4][4]; // [mtile][ntile], wave owns d-cols [wave*64, wave*64+64)
#pragma unroll
    for (int m = 0; m < 4; ++m)
#pragma unroll
        for (int n = 0; n < 4; ++n)
            yacc[m][n] = (v8f)(0.f);

    const int ffbase0 = part * 1024;

    for (int ch = 0; ch < 1024 / FCH; ++ch) {
        const int fb   = ffbase0 + ch * FCH;
        const int fcol = fb + wave * 16 + col;  // this lane's B column, phase 1

        v8f g[4], u[4];
#pragma unroll
        for (int m = 0; m < 4; ++m) { g[m] = (v8f)(0.f); u[m] = (v8f)(0.f); }

        for (int k = 0; k < DMODEL; k += 32) {
            BFrag a[4], bg, bu;
            const int aoff = k + hi * 8;
#pragma unroll
            for (int m = 0; m < 4; ++m) {
                a[m].q[0] = *(const uint4*)(arow[m] + aoff);
                a[m].q[1] = *(const uint4*)(arow[m] + aoff + 16);
            }
#pragma unroll
            for (int j = 0; j < 8; ++j) {
                const size_t k1 = (size_t)(k + hi * 8 + j)      * DFF + fcol;
                const size_t k2 = (size_t)(k + 16 + hi * 8 + j) * DFF + fcol;
                bg.b[j]     = f2bf(wg[k1]);
                bg.b[8 + j] = f2bf(wg[k2]);
                bu.b[j]     = f2bf(wu[k1]);
                bu.b[8 + j] = f2bf(wu[k2]);
            }
            // each B fragment amortized over 4 M-tiles
#pragma unroll
            for (int m = 0; m < 4; ++m)
                g[m] = __builtin_amdgcn_wmma_f32_16x16x32_bf16(
                    false, a[m].v, false, bg.v, (short)0, g[m], false, false);
#pragma unroll
            for (int m = 0; m < 4; ++m)
                u[m] = __builtin_amdgcn_wmma_f32_16x16x32_bf16(
                    false, a[m].v, false, bu.v, (short)0, u[m], false, false);
        }
        const float bgv = b_gate[e * DFF + fcol];
        const float buv = b_up[e * DFF + fcol];
#pragma unroll
        for (int m = 0; m < 4; ++m)
#pragma unroll
            for (int r = 0; r < 8; ++r) {
                float h = silu(g[m][r] + bgv) * (u[m][r] + buv);
                sh[(m * 16 + r + 8 * hi) * SHSTR + wave * 16 + col] = f2bf(h);
            }
        __syncthreads();

        // phase 2: y[64, wave's 64 cols] += h[64,FCH] @ Wd[fb:fb+FCH, :]
#pragma unroll
        for (int kk = 0; kk < FCH; kk += 32) {
            BFrag am[4];
#pragma unroll
            for (int m = 0; m < 4; ++m) {
                const int off = (m * 16 + col) * SHSTR + kk + hi * 8;
                am[m].q[0] = *(const uint4*)(sh + off);
                am[m].q[1] = *(const uint4*)(sh + off + 16);
            }
#pragma unroll
            for (int nt = 0; nt < 4; ++nt) {
                const int dcol = wave * 64 + nt * 16 + col;
                BFrag bd;
#pragma unroll
                for (int j = 0; j < 8; ++j) {
                    const size_t f1 = (size_t)(fb + kk + hi * 8 + j)      * DMODEL + dcol;
                    const size_t f2 = (size_t)(fb + kk + 16 + hi * 8 + j) * DMODEL + dcol;
                    bd.b[j]     = f2bf(wd[f1]);
                    bd.b[8 + j] = f2bf(wd[f2]);
                }
#pragma unroll
                for (int m = 0; m < 4; ++m)
                    yacc[m][nt] = __builtin_amdgcn_wmma_f32_16x16x32_bf16(
                        false, am[m].v, false, bd.v, (short)0, yacc[m][nt], false, false);
            }
        }
        __syncthreads();
    }

    // epilogue: combine with gate weights, scatter-add into output
#pragma unroll
    for (int nt = 0; nt < 4; ++nt) {
        const int dcol = wave * 64 + nt * 16 + col;
        const float bdv = b_down[e * DMODEL + dcol];
#pragma unroll
        for (int m = 0; m < 4; ++m)
#pragma unroll
            for (int r = 0; r < 8; ++r) {
                const int row = m * 16 + r + 8 * hi;
                const float val = (yacc[m][nt][r] + bdv) * swt[row];
                unsafeAtomicAdd(&out[(size_t)stok[row] * DMODEL + dcol], val);
            }
    }
}

// ---------------------------------------------------------------------------
extern "C" void kernel_launch(void* const* d_in, const int* in_sizes, int n_in,
                              void* d_out, int out_size, void* d_ws, size_t ws_size,
                              hipStream_t stream) {
    const float* x      = (const float*)d_in[0];
    const float* Wg     = (const float*)d_in[1];
    const float* w_gate = (const float*)d_in[2];
    const float* b_gate = (const float*)d_in[3];
    const float* w_up   = (const float*)d_in[4];
    const float* b_up   = (const float*)d_in[5];
    const float* w_down = (const float*)d_in[6];
    const float* b_down = (const float*)d_in[7];
    float* out = (float*)d_out;

    char* ws = (char*)d_ws;
    int*    counts = (int*)(ws + 0);
    int*    list   = (int*)(ws + 4096);
    float*  wlist  = (float*)(ws + 4096 + NEXP * NTOK * 4);
    __bf16* xbf    = (__bf16*)(ws + 4096 + 2 * NEXP * NTOK * 4);

    hipMemsetAsync(out, 0, (size_t)out_size * sizeof(float), stream);
    hipMemsetAsync(counts, 0, NEXP * sizeof(int), stream);

    moe_gate_kernel<<<NTOK / 8, 256, 0, stream>>>(x, Wg, counts, list, wlist, xbf);

    dim3 grid(NTOK / 64, 4, NEXP);   // (mtile, dff partition, expert)
    moe_ffn_kernel<<<grid, 512, 0, stream>>>(xbf, counts, list, wlist,
                                             w_gate, b_gate, w_up, b_up,
                                             w_down, b_down, out);
}
